// LSTMClassifier_62715112456547
// MI455X (gfx1250) — compile-verified, matrix operand order checked by reading, not run on hardware
//
#include <hip/hip_runtime.h>
#include <hip/hip_bf16.h>
#include <math.h>

typedef __attribute__((ext_vector_type(16))) __bf16 v16bf;
typedef __attribute__((ext_vector_type(8)))  float  v8f;

#define T_SEQ 4096
#define I_DIM 1024
#define H_DIM 2048
#define O_DIM 64

#define BK        32      // K panel depth
#define LDSTRIDE  40      // halves per LDS row: 32 data + 8 pad (80B, conflict-free b128)
#define NKSTEPS   (I_DIM / BK)   // 32

#if defined(__HIP_DEVICE_COMPILE__) && __has_builtin(__builtin_amdgcn_tensor_load_to_lds) && __has_builtin(__builtin_amdgcn_s_wait_tensorcnt)
#define USE_TDM 1
#else
#define USE_TDM 0
#endif

#if USE_TDM
typedef __attribute__((ext_vector_type(4))) unsigned int v4u;
typedef __attribute__((ext_vector_type(8))) int          v8i;
typedef __attribute__((ext_vector_type(4))) int          v4i;

// 2D TDM load: tile (32 x rows) of bf16 from row-major [tensor_rows, tensor_cols],
// written to LDS with 16B padding after every 64B row -> LDSTRIDE layout.
__device__ __forceinline__ void tdm_load_2d(unsigned lds_off, const void* gptr,
                                            unsigned rows, unsigned tensor_rows,
                                            unsigned tensor_cols) {
  unsigned long long ga = (unsigned long long)(size_t)gptr;
  union { unsigned int w[4]; v4u v; } g0 = {};
  g0.w[0] = 1u;                                             // count=1, no gather
  g0.w[1] = lds_off;                                        // lds_addr [63:32]
  g0.w[2] = (unsigned)(ga & 0xFFFFFFFFu);                   // global_addr [95:64]
  g0.w[3] = (unsigned)((ga >> 32) & 0x01FFFFFFu)            // global_addr [120:96]
          | (2u << 30);                                     // type=2 (image)
  union { int w[8]; v8i v; } g1 = {};
  g1.w[0] = (int)((1u << 16)     // data_size = 2 bytes
                | (1u << 20)     // pad_enable
                | (3u << 22)     // pad_interval: every 16 DWORDs (64B)
                | (3u << 25));   // pad_amount: 4 DWORDs (16B)
  g1.w[1] = (int)((tensor_cols & 0xFFFFu) << 16);           // tensor_dim0[15:0] @ bits[63:48]
  g1.w[2] = (int)(((tensor_cols >> 16) & 0xFFFFu)           // tensor_dim0[31:16]
                | ((tensor_rows & 0xFFFFu) << 16));         // tensor_dim1[15:0]
  g1.w[3] = (int)(((tensor_rows >> 16) & 0xFFFFu)           // tensor_dim1[31:16]
                | (32u << 16));                             // tile_dim0 = 32
  g1.w[4] = (int)(rows & 0xFFFFu);                          // tile_dim1 (tile_dim2 = 0)
  g1.w[5] = (int)tensor_cols;                               // tensor_dim0_stride[31:0]
  g1.w[6] = 0;                                              // stride hi / dim1_stride lo
  g1.w[7] = 0;
  v4i z4 = {};
  v8i z8 = {};
  // 6-arg form (clang-23 / therock-10.0): (v4u g0, v8i g1, v4i g2, v4i g3, v8i extra, i32 cpol)
  __builtin_amdgcn_tensor_load_to_lds(g0.v, g1.v, z4, z4, z8, 0);
}
#endif

// ---------------- f32 -> bf16 (round to nearest even) ----------------
__device__ __forceinline__ unsigned short f32_to_bf16(float x) {
  unsigned int u = __float_as_uint(x);
  unsigned int lsb = (u >> 16) & 1u;
  u += 0x7FFFu + lsb;
  return (unsigned short)(u >> 16);
}

__global__ void cvt_bf16_kernel(const float* __restrict__ src,
                                unsigned short* __restrict__ dst, int n) {
  int i = blockIdx.x * blockDim.x + threadIdx.x;
  if (i < n) dst[i] = f32_to_bf16(src[i]);
}

// ---------------- zero-init h double-buffer + barrier ----------------
__global__ void init_kernel(float* __restrict__ hbuf, unsigned int* __restrict__ bar) {
  int i = blockIdx.x * blockDim.x + threadIdx.x;
  if (i < 2 * H_DIM) hbuf[i] = 0.0f;
  if (i == 0) *bar = 0u;
}

// ---------------- fallback panel copy (global -> padded LDS) ----------------
__device__ __forceinline__ void copy_panel(const unsigned short* __restrict__ g,
                                           unsigned short* s, int rows, int tid) {
  // rows x 32 halves; 4x 16B chunks per row
  for (int c = tid; c < rows * 4; c += 256) {
    int row = c >> 2, ch = c & 3;
    *(uint4*)(s + row * LDSTRIDE + ch * 8) = *(const uint4*)(g + (size_t)row * I_DIM + ch * 8);
  }
}

// ---------------- Phase 1: gate pre-activation GEMMs ----------------
// Block (256 thr, 8 waves) computes a 64x128 tile of xg[g] = X * W_g^T + b_g.
// Panels staged in LDS (TDM double-buffered); each wave: 32x32 register block.
__global__ void __launch_bounds__(256)
gate_gemm_kernel(const unsigned short* __restrict__ Xbf,   // [T, I] bf16
                 const unsigned short* __restrict__ Wbf,   // [4][H, I] bf16
                 const float* __restrict__ b0, const float* __restrict__ b1,
                 const float* __restrict__ b2, const float* __restrict__ b3,
                 float* __restrict__ xg)                   // [4][T, H] f32
{
  __shared__ __align__(16) unsigned short Apan[2][64 * LDSTRIDE];    // 2 x 5120 B
  __shared__ __align__(16) unsigned short Bpan[2][128 * LDSTRIDE];   // 2 x 10240 B

  const int tid  = threadIdx.x;
  const int lane = tid & 31;
  const int wave = tid >> 5;
  const int r  = lane & 15;
  const int hi = lane >> 4;
  const int wm = wave >> 2;       // 0..1  (M sub-block)
  const int wn = wave & 3;        // 0..3  (N sub-block)

  const int nTilesN = H_DIM / 128;                       // 16
  const int blocksPerGate = (T_SEQ / 64) * nTilesN;      // 1024
  const int gate = blockIdx.x / blocksPerGate;
  const int rem  = blockIdx.x % blocksPerGate;
  const int tm = (rem / nTilesN) * 64;
  const int tn = (rem % nTilesN) * 128;

  const unsigned short* Abase = Xbf + (size_t)tm * I_DIM;
  const unsigned short* Bbase = Wbf + ((size_t)gate * H_DIM + (size_t)tn) * I_DIM;

  v8f acc[2][2] = {{{}, {}}, {{}, {}}};

#if USE_TDM
  const unsigned ldsA0 = (unsigned)(size_t)&Apan[0][0];
  const unsigned ldsA1 = (unsigned)(size_t)&Apan[1][0];
  const unsigned ldsB0 = (unsigned)(size_t)&Bpan[0][0];
  const unsigned ldsB1 = (unsigned)(size_t)&Bpan[1][0];
  if (wave == 0) {
    tdm_load_2d(ldsA0, Abase, 64, T_SEQ, I_DIM);
    tdm_load_2d(ldsB0, Bbase, 128, H_DIM, I_DIM);
  }
#else
  copy_panel(Abase, &Apan[0][0], 64, tid);
  copy_panel(Bbase, &Bpan[0][0], 128, tid);
#endif

  for (int kk = 0; kk < NKSTEPS; ++kk) {
    const int cur = kk & 1;
#if USE_TDM
    if (wave == 0) {
      if (kk + 1 < NKSTEPS) {
        const int nxt = (kk + 1) & 1;
        const int k0n = (kk + 1) * BK;
        tdm_load_2d(nxt ? ldsA1 : ldsA0, Abase + k0n, 64, T_SEQ, I_DIM);
        tdm_load_2d(nxt ? ldsB1 : ldsB0, Bbase + k0n, 128, H_DIM, I_DIM);
        __builtin_amdgcn_s_wait_tensorcnt((short)2);  // pair for buf `cur` done
      } else {
        __builtin_amdgcn_s_wait_tensorcnt((short)0);
      }
    }
#else
    if (kk + 1 < NKSTEPS) {
      const int nxt = (kk + 1) & 1;
      const int k0n = (kk + 1) * BK;
      copy_panel(Abase + k0n, &Apan[nxt][0], 64, tid);
      copy_panel(Bbase + k0n, &Bpan[nxt][0], 128, tid);
    }
#endif
    __syncthreads();   // buffer `cur` ready (and prev compute finished)

    const unsigned short* Ap = &Apan[cur][0];
    const unsigned short* Bp = &Bpan[cur][0];

    union { uint4 q[2]; v16bf v; } afrag[2], bfrag[2];
#pragma unroll
    for (int mi = 0; mi < 2; ++mi) {
      const int row = wm * 32 + mi * 16 + r;
      afrag[mi].q[0] = *(const uint4*)(Ap + row * LDSTRIDE + 8 * hi);
      afrag[mi].q[1] = *(const uint4*)(Ap + row * LDSTRIDE + 16 + 8 * hi);
    }
#pragma unroll
    for (int ni = 0; ni < 2; ++ni) {
      const int row = wn * 32 + ni * 16 + r;
      bfrag[ni].q[0] = *(const uint4*)(Bp + row * LDSTRIDE + 8 * hi);
      bfrag[ni].q[1] = *(const uint4*)(Bp + row * LDSTRIDE + 16 + 8 * hi);
    }
#pragma unroll
    for (int mi = 0; mi < 2; ++mi)
#pragma unroll
      for (int ni = 0; ni < 2; ++ni)
        acc[mi][ni] = __builtin_amdgcn_wmma_f32_16x16x32_bf16(
            false, afrag[mi].v, false, bfrag[ni].v,
            (short)0, acc[mi][ni], false, false);

    __syncthreads();   // all waves done with buffer `cur` before TDM overwrites it
  }

  const float* bias = (gate == 0) ? b0 : (gate == 1) ? b1 : (gate == 2) ? b2 : b3;
  float* outg = xg + (size_t)gate * T_SEQ * H_DIM;
#pragma unroll
  for (int mi = 0; mi < 2; ++mi) {
#pragma unroll
    for (int ni = 0; ni < 2; ++ni) {
      const int col = tn + wn * 32 + ni * 16 + r;
      const float bv = bias[col];
#pragma unroll
      for (int i = 0; i < 8; ++i) {
        const int row = tm + wm * 32 + mi * 16 + i + 8 * hi;
        outg[(size_t)row * H_DIM + col] = acc[mi][ni][i] + bv;
      }
    }
  }
}

// ---------------- Phase 2: sequential LSTM scan (persistent, grid barrier) ----------------
__device__ __forceinline__ float bf16_lo(unsigned int u) { return __uint_as_float(u << 16); }
__device__ __forceinline__ float bf16_hi(unsigned int u) { return __uint_as_float(u & 0xFFFF0000u); }

__global__ void __launch_bounds__(256)
lstm_scan_kernel(const unsigned short* __restrict__ Ubf,  // [4][H][H] bf16
                 const float* __restrict__ xg,            // [4][T][H] f32
                 const float* __restrict__ Ufb, const float* __restrict__ Uib,
                 const float* __restrict__ Ucb, const float* __restrict__ Uob,
                 float* __restrict__ hbuf,                // [2][H]
                 unsigned int* __restrict__ bar)
{
  __shared__ float h_lds[H_DIM];      // 8 KB
  __shared__ float partial[64][4];
  __shared__ float c_lds[16];

  const int tid   = threadIdx.x;
  const int jbase = blockIdx.x * 16;

  if (tid < 16) c_lds[tid] = 0.0f;

  const int dp   = tid >> 2;          // 0..63 : gate = dp>>4, row jj = dp&15
  const int quad = tid & 3;           // 0..3  : K segment
  const int g    = dp >> 4;
  const int jj   = dp & 15;
  const unsigned short* Urow =
      Ubf + ((size_t)g * H_DIM + (size_t)(jbase + jj)) * (size_t)H_DIM + quad * 512;

  for (int t = 0; t < T_SEQ; ++t) {
    const float* hsrc = hbuf + (t & 1) * H_DIM;
    __syncthreads();
    for (int idx = tid; idx < H_DIM; idx += 256) h_lds[idx] = hsrc[idx];
    __syncthreads();

    float s = 0.0f;
    const float* hseg = h_lds + quad * 512;
    for (int k = 0; k < 512; k += 8) {
      uint4 q = *(const uint4*)(Urow + k);
      s = fmaf(bf16_lo(q.x), hseg[k + 0], s);
      s = fmaf(bf16_hi(q.x), hseg[k + 1], s);
      s = fmaf(bf16_lo(q.y), hseg[k + 2], s);
      s = fmaf(bf16_hi(q.y), hseg[k + 3], s);
      s = fmaf(bf16_lo(q.z), hseg[k + 4], s);
      s = fmaf(bf16_hi(q.z), hseg[k + 5], s);
      s = fmaf(bf16_lo(q.w), hseg[k + 6], s);
      s = fmaf(bf16_hi(q.w), hseg[k + 7], s);
    }
    partial[dp][quad] = s;
    __syncthreads();

    if (tid < 16) {
      const int j = jbase + tid;
      float df = partial[0 * 16 + tid][0] + partial[0 * 16 + tid][1] +
                 partial[0 * 16 + tid][2] + partial[0 * 16 + tid][3];
      float di = partial[1 * 16 + tid][0] + partial[1 * 16 + tid][1] +
                 partial[1 * 16 + tid][2] + partial[1 * 16 + tid][3];
      float dc = partial[2 * 16 + tid][0] + partial[2 * 16 + tid][1] +
                 partial[2 * 16 + tid][2] + partial[2 * 16 + tid][3];
      float dq = partial[3 * 16 + tid][0] + partial[3 * 16 + tid][1] +
                 partial[3 * 16 + tid][2] + partial[3 * 16 + tid][3];

      float af = xg[((size_t)0 * T_SEQ + t) * H_DIM + j] + Ufb[j] + df;
      float ai = xg[((size_t)1 * T_SEQ + t) * H_DIM + j] + Uib[j] + di;
      float ac = xg[((size_t)2 * T_SEQ + t) * H_DIM + j] + Ucb[j] + dc;
      float ao = xg[((size_t)3 * T_SEQ + t) * H_DIM + j] + Uob[j] + dq;

      float f  = 1.0f / (1.0f + expf(-af));
      float ii = 1.0f / (1.0f + expf(-ai));
      float ch = tanhf(ac);
      float oo = 1.0f / (1.0f + expf(-ao));   // uses previous h, as in reference

      float cn = f * c_lds[tid] + ii * ch;
      c_lds[tid] = cn;
      hbuf[((t + 1) & 1) * H_DIM + j] = oo * tanhf(cn);
    }
    __syncthreads();

    if (tid == 0) {
      __threadfence();
      atomicAdd(bar, 1u);
      unsigned int target = (unsigned int)(t + 1) * gridDim.x;
      while (atomicAdd(bar, 0u) < target) { __builtin_amdgcn_s_sleep(1); }
      __threadfence();
    }
    __syncthreads();
  }
}

// ---------------- Phase 3: FC + log_softmax ----------------
__global__ void __launch_bounds__(256)
final_kernel(const float* __restrict__ h,        // final h (buf 0), [H]
             const float* __restrict__ fc_w,     // [64, H]
             const float* __restrict__ fc_b,     // [64]
             float* __restrict__ out)            // [64]
{
  __shared__ float partial[64][4];
  __shared__ float logits[64];
  __shared__ float m_s, l_s;

  const int tid = threadIdx.x;
  const int o = tid >> 2, quad = tid & 3;
  const float* wrow = fc_w + (size_t)o * H_DIM + quad * 512;
  const float* hseg = h + quad * 512;
  float s = 0.0f;
  for (int k = 0; k < 512; ++k) s = fmaf(wrow[k], hseg[k], s);
  partial[o][quad] = s;
  __syncthreads();
  if (tid < 64) {
    logits[tid] = partial[tid][0] + partial[tid][1] + partial[tid][2] + partial[tid][3]
                + fc_b[tid];
  }
  __syncthreads();
  if (tid == 0) {
    float m = logits[0];
    for (int i = 1; i < O_DIM; ++i) m = fmaxf(m, logits[i]);
    float sum = 0.0f;
    for (int i = 0; i < O_DIM; ++i) sum += expf(logits[i] - m);
    m_s = m; l_s = logf(sum);
  }
  __syncthreads();
  if (tid < 64) out[tid] = logits[tid] - m_s - l_s;
}

// ---------------- launcher ----------------
extern "C" void kernel_launch(void* const* d_in, const int* in_sizes, int n_in,
                              void* d_out, int out_size, void* d_ws, size_t ws_size,
                              hipStream_t stream) {
  (void)in_sizes; (void)n_in; (void)out_size; (void)ws_size;

  const float* input_seq = (const float*)d_in[0];
  const float* Ww[4] = { (const float*)d_in[1], (const float*)d_in[5],
                         (const float*)d_in[9], (const float*)d_in[13] };
  const float* Wb[4] = { (const float*)d_in[2], (const float*)d_in[6],
                         (const float*)d_in[10], (const float*)d_in[14] };
  const float* Uw[4] = { (const float*)d_in[3], (const float*)d_in[7],
                         (const float*)d_in[11], (const float*)d_in[15] };
  const float* Ub[4] = { (const float*)d_in[4], (const float*)d_in[8],
                         (const float*)d_in[12], (const float*)d_in[16] };
  const float* fc_w = (const float*)d_in[17];
  const float* fc_b = (const float*)d_in[18];
  float* out = (float*)d_out;

  // workspace layout (256B aligned)
  char* ws = (char*)d_ws;
  const size_t XBF_OFF = 0;                                       // T*I bf16    = 8 MB
  const size_t WBF_OFF = XBF_OFF + (size_t)T_SEQ * I_DIM * 2;     // 4*H*I bf16  = 16 MB
  const size_t UBF_OFF = WBF_OFF + (size_t)4 * H_DIM * I_DIM * 2; // 4*H*H bf16  = 33.5 MB
  const size_t XG_OFF  = UBF_OFF + (size_t)4 * H_DIM * H_DIM * 2; // 4*T*H f32   = 134 MB
  const size_t H_OFF   = XG_OFF + (size_t)4 * T_SEQ * H_DIM * 4;  // 2*H f32
  const size_t BAR_OFF = (H_OFF + (size_t)2 * H_DIM * 4 + 255) & ~(size_t)255;

  unsigned short* Xbf = (unsigned short*)(ws + XBF_OFF);
  unsigned short* Wbf = (unsigned short*)(ws + WBF_OFF);
  unsigned short* Ubf = (unsigned short*)(ws + UBF_OFF);
  float*          xg  = (float*)(ws + XG_OFF);
  float*          hb  = (float*)(ws + H_OFF);
  unsigned int*   bar = (unsigned int*)(ws + BAR_OFF);

  // 0) precision conversion: f32 -> bf16 (X, W[4], U[4])
  {
    int n = T_SEQ * I_DIM;
    cvt_bf16_kernel<<<(n + 255) / 256, 256, 0, stream>>>(input_seq, Xbf, n);
    n = H_DIM * I_DIM;
    for (int g = 0; g < 4; ++g)
      cvt_bf16_kernel<<<(n + 255) / 256, 256, 0, stream>>>(
          Ww[g], Wbf + (size_t)g * H_DIM * I_DIM, n);
    n = H_DIM * H_DIM;
    for (int g = 0; g < 4; ++g)
      cvt_bf16_kernel<<<(n + 255) / 256, 256, 0, stream>>>(
          Uw[g], Ubf + (size_t)g * H_DIM * H_DIM, n);
  }

  // 1) gate pre-activations: LDS-tiled WMMA GEMM, TDM double-buffered panels
  {
    const int blocks = 4 * (T_SEQ / 64) * (H_DIM / 128);   // 4096
    gate_gemm_kernel<<<blocks, 256, 0, stream>>>(Xbf, Wbf, Wb[0], Wb[1], Wb[2], Wb[3], xg);
  }

  // 2) init h/c/barrier, then persistent sequential scan
  init_kernel<<<(2 * H_DIM + 255) / 256, 256, 0, stream>>>(hb, bar);
  lstm_scan_kernel<<<H_DIM / 16, 256, 0, stream>>>(Ubf, xg, Ub[0], Ub[1], Ub[2], Ub[3], hb, bar);

  // 3) FC + log_softmax  (final h lives in buffer 0 after 4096 steps)
  final_kernel<<<1, 256, 0, stream>>>(hb, fc_w, fc_b, out);
}
